// EncoderTransformerBlock_71296457114377
// MI455X (gfx1250) — compile-verified, matrix operand order checked by reading, not run on hardware
//
#include <hip/hip_runtime.h>

// ---------------------------------------------------------------------------
// Swin-style EncoderTransformerBlock for MI455X (gfx1250, wave32, WMMA).
// bf16 WMMA (v_wmma_f32_16x16x32_bf16) for all large GEMMs + attention,
// fp32 VALU for softmax / LN / 8x8 FFTs / 40-token Fourier attention.
// ---------------------------------------------------------------------------

typedef __bf16 bf16;
typedef bf16  bf16x8  __attribute__((ext_vector_type(8)));
typedef bf16  bf16x16 __attribute__((ext_vector_type(16)));
typedef float f32x8   __attribute__((ext_vector_type(8)));

#define WMMA_BF16(a, b, c) \
  __builtin_amdgcn_wmma_f32_16x16x32_bf16(false, (a), false, (b), (short)0, (c), false, false)

// Problem constants
#define Bv 8
#define Hv 128
#define Wv 128
#define Cv 192
#define WSv 8
#define SSv 4
#define HEADSv 6
#define NWv 256           // windows per image
#define Nv 64             // tokens per window
#define BWv 2048          // total windows
#define MTOK 131072       // B*H*W tokens
#define MFOUR 163840      // BWv * 80 (40 freq rows, real+imag)
#define HDv 32
#define SCALEv 0.17677669529663687f   // 32^-0.5

__constant__ float COS8[8] = {1.f, 0.70710678118654752f, 0.f, -0.70710678118654752f,
                              -1.f, -0.70710678118654752f, 0.f, 0.70710678118654752f};
__constant__ float SIN8[8] = {0.f, 0.70710678118654752f, 1.f, 0.70710678118654752f,
                              0.f, -0.70710678118654752f, -1.f, -0.70710678118654752f};

__device__ __forceinline__ bf16 f2bf(float f) {
  union { float f; unsigned u; } in; in.f = f;
  unsigned r = in.u + 0x7FFFu + ((in.u >> 16) & 1u);   // RTNE
  union { unsigned short s; bf16 b; } out; out.s = (unsigned short)(r >> 16);
  return out.b;
}

// A/B fragment (16-bit, 16x32): lane L<16 -> row L, k in {kh*8..kh*8+7, 16+kh*8..}
// where kh = lane>>4.  Two contiguous 16B chunks per lane.
__device__ __forceinline__ bf16x16 load_frag(const bf16* __restrict__ p, int ld) {
  int lane = threadIdx.x & 31;
  int row = lane & 15, kh = lane >> 4;
  const bf16* r0 = p + (size_t)row * ld + kh * 8;
  bf16x8 a = *(const bf16x8*)(r0);
  bf16x8 b = *(const bf16x8*)(r0 + 16);
  bf16x16 f;
#pragma unroll
  for (int i = 0; i < 8; i++) { f[i] = a[i]; f[i + 8] = b[i]; }
  return f;
}

// B fragment loaded from an UN-transposed (K x ncols) matrix: lane holds column
// (lane&15); k elements strided by ld.
__device__ __forceinline__ bf16x16 load_fragT(const bf16* __restrict__ p, int ld) {
  int lane = threadIdx.x & 31;
  int col = lane & 15, kh = lane >> 4;
  const bf16* base = p + col + (size_t)(kh * 8) * ld;
  bf16x16 f;
#pragma unroll
  for (int e = 0; e < 8; e++) {
    f[e]     = base[(size_t)e * ld];
    f[e + 8] = base[(size_t)(e + 16) * ld];
  }
  return f;
}

__device__ __forceinline__ float wave_sum(float v) {
  for (int m = 16; m; m >>= 1) v += __shfl_xor(v, m, 32);
  return v;
}

// ---------------------------------------------------------------------------
// Weight transpose (K,N) fp32 -> (N,K) bf16 so B fragments load like A frags.
// ---------------------------------------------------------------------------
__global__ void transpose_to_bf16(const float* __restrict__ w, bf16* __restrict__ wt,
                                  int K, int N) {
  int i = blockIdx.x * blockDim.x + threadIdx.x;
  if (i >= K * N) return;
  int k = i / N, n = i % N;
  wt[(size_t)n * K + k] = f2bf(w[i]);
}

// ---------------------------------------------------------------------------
// LN1 + cyclic shift + window partition.  One wave per destination token.
// dest row m = win*64 + n; source = xn[(hs+4)&127][(ws+4)&127].
// ---------------------------------------------------------------------------
__global__ __launch_bounds__(256) void ln_shift_partition(
    const float* __restrict__ x, const float* __restrict__ g, const float* __restrict__ bta,
    float* __restrict__ xw, bf16* __restrict__ xwb) {
  int wid = (blockIdx.x * blockDim.x + threadIdx.x) >> 5;
  int lane = threadIdx.x & 31;
  if (wid >= MTOK) return;
  int win = wid >> 6, n = wid & 63;
  int bb = win >> 8, wi = win & 255, wh = wi >> 4, ww = wi & 15;
  int r = n >> 3, cw = n & 7;
  int h = (wh * 8 + r + SSv) & 127;
  int w = (ww * 8 + cw + SSv) & 127;
  const float* src = x + ((size_t)((bb * Hv + h) * Wv + w)) * Cv;
  float v[6]; float s = 0.f, s2 = 0.f;
#pragma unroll
  for (int j = 0; j < 6; j++) {
    int c = lane + j * 32;
    v[j] = src[c]; s += v[j]; s2 += v[j] * v[j];
  }
  s = wave_sum(s); s2 = wave_sum(s2);
  float mean = s * (1.f / Cv);
  float var = s2 * (1.f / Cv) - mean * mean;
  float rs = rsqrtf(var + 1e-5f);
#pragma unroll
  for (int j = 0; j < 6; j++) {
    int c = lane + j * 32;
    float y = (v[j] - mean) * rs * g[c] + bta[c];
    xw[(size_t)wid * Cv + c] = y;
    xwb[(size_t)wid * Cv + c] = f2bf(y);
  }
}

// LN2 over x1 rows (already in (B,H,W,C) order).
__global__ __launch_bounds__(256) void ln2_kernel(
    const float* __restrict__ x1, const float* __restrict__ g, const float* __restrict__ bta,
    bf16* __restrict__ yn) {
  int wid = (blockIdx.x * blockDim.x + threadIdx.x) >> 5;
  int lane = threadIdx.x & 31;
  if (wid >= MTOK) return;
  const float* src = x1 + (size_t)wid * Cv;
  float v[6]; float s = 0.f, s2 = 0.f;
#pragma unroll
  for (int j = 0; j < 6; j++) {
    int c = lane + j * 32;
    v[j] = src[c]; s += v[j]; s2 += v[j] * v[j];
  }
  s = wave_sum(s); s2 = wave_sum(s2);
  float mean = s * (1.f / Cv);
  float var = s2 * (1.f / Cv) - mean * mean;
  float rs = rsqrtf(var + 1e-5f);
#pragma unroll
  for (int j = 0; j < 6; j++) {
    int c = lane + j * 32;
    yn[(size_t)wid * Cv + c] = f2bf((v[j] - mean) * rs * g[c] + bta[c]);
  }
}

// ---------------------------------------------------------------------------
// Generic bf16 WMMA GEMM: (MxK) @ (KxN) with B pre-transposed to (N,K).
// Block = 256 threads (8 waves); block tile 128(M) x 64(N); wave tile 32x32.
// EPI: 0 = bias->bf16; 1 = bias+GELU->bf16;
//      2 = bias + window-reverse/unshift + shortcut residual -> fp32 (proj);
//      3 = bias + x1 residual -> fp32 (fc2, final output).
// ---------------------------------------------------------------------------
template <int EPI>
__global__ __launch_bounds__(256) void gemm_bf16(
    const bf16* __restrict__ A, const bf16* __restrict__ Bt,
    const float* __restrict__ bias, void* __restrict__ outp,
    const float* __restrict__ aux, int M, int N, int K) {
  int lane = threadIdx.x & 31;
  int wave = threadIdx.x >> 5;
  int m0 = blockIdx.x * 128 + (wave & 3) * 32;
  int n0 = blockIdx.y * 64 + (wave >> 2) * 32;
  f32x8 acc[2][2] = {};
  for (int kb = 0; kb < K; kb += 32) {
    bf16x16 a0 = load_frag(A + (size_t)m0 * K + kb, K);
    bf16x16 a1 = load_frag(A + (size_t)(m0 + 16) * K + kb, K);
    bf16x16 b0 = load_frag(Bt + (size_t)n0 * K + kb, K);
    bf16x16 b1 = load_frag(Bt + (size_t)(n0 + 16) * K + kb, K);
    acc[0][0] = WMMA_BF16(a0, b0, acc[0][0]);
    acc[0][1] = WMMA_BF16(a0, b1, acc[0][1]);
    acc[1][0] = WMMA_BF16(a1, b0, acc[1][0]);
    acc[1][1] = WMMA_BF16(a1, b1, acc[1][1]);
  }
  int ln = lane & 15, mh = lane >> 4;
#pragma unroll
  for (int i = 0; i < 2; i++)
#pragma unroll
    for (int j = 0; j < 2; j++)
#pragma unroll
      for (int r = 0; r < 8; r++) {
        int m = m0 + i * 16 + mh * 8 + r;
        int n = n0 + j * 16 + ln;
        float val = acc[i][j][r] + bias[n];
        if (EPI == 0) {
          ((bf16*)outp)[(size_t)m * N + n] = f2bf(val);
        } else if (EPI == 1) {
          float gv = 0.5f * val * (1.f + erff(val * 0.70710678118654752f));
          ((bf16*)outp)[(size_t)m * N + n] = f2bf(gv);
        } else if (EPI == 2) {
          int win = m >> 6, nn = m & 63;
          int bb = win >> 8, wi = win & 255, wh = wi >> 4, ww = wi & 15;
          int rr = nn >> 3, cw = nn & 7;
          int hh = (wh * 8 + rr + SSv) & 127;     // unshift
          int w2 = (ww * 8 + cw + SSv) & 127;
          size_t orow = (size_t)((bb * Hv + hh) * Wv + w2);
          ((float*)outp)[orow * Cv + n] = val + aux[orow * Cv + n];
        } else {
          ((float*)outp)[(size_t)m * Cv + n] = val + aux[(size_t)m * Cv + n];
        }
      }
}

// ---------------------------------------------------------------------------
// Spatial window attention.  One block per window; wave w (w<6) owns head w.
// scores = WMMA(q,k) * scale + rel_bias + mask; fp32 softmax over 64 cols
// (4 n-tiles x 16 lanes, __shfl_xor reduce); probs -> LDS bf16; P@V via WMMA.
// ---------------------------------------------------------------------------
__global__ __launch_bounds__(256) void spatial_attn(
    const bf16* __restrict__ qkv, const float* __restrict__ mask,
    const float* __restrict__ relt, float* __restrict__ xsp) {
  __shared__ bf16 ldsP[HEADSv][Nv * Nv];   // 48 KB
  int win = blockIdx.x;
  int wave = threadIdx.x >> 5;
  int lane = threadIdx.x & 31;
  if (wave >= HEADSv) return;              // no __syncthreads in this kernel
  int h = wave;
  const bf16* q = qkv + (size_t)win * Nv * 576 + h * HDv;
  const bf16* k = q + Cv;
  const bf16* v = q + 2 * Cv;
  int ln = lane & 15, mh = lane >> 4;
  const float* mrow = mask + (size_t)(win & 255) * (Nv * Nv);

  for (int mt = 0; mt < 4; mt++) {
    bf16x16 a = load_frag(q + (size_t)(mt * 16) * 576, 576);
    f32x8 sc[4];
#pragma unroll
    for (int nt = 0; nt < 4; nt++) {
      bf16x16 bfr = load_frag(k + (size_t)(nt * 16) * 576, 576);
      f32x8 z = {};
      sc[nt] = WMMA_BF16(a, bfr, z);
    }
#pragma unroll
    for (int r = 0; r < 8; r++) {
      int m = mt * 16 + mh * 8 + r;
      int mr = m >> 3, mc = m & 7;
      float vals[4]; float mx = -1e30f;
#pragma unroll
      for (int nt = 0; nt < 4; nt++) {
        int n = nt * 16 + ln;
        int nr = n >> 3, nc = n & 7;
        int idx = (mr - nr + 7) * 15 + (mc - nc + 7);
        float s = sc[nt][r] * SCALEv + relt[idx * HEADSv + h] + mrow[m * Nv + n];
        vals[nt] = s; mx = fmaxf(mx, s);
      }
      for (int d = 1; d < 16; d <<= 1) mx = fmaxf(mx, __shfl_xor(mx, d, 32));
      float se = 0.f;
#pragma unroll
      for (int nt = 0; nt < 4; nt++) { vals[nt] = __expf(vals[nt] - mx); se += vals[nt]; }
      for (int d = 1; d < 16; d <<= 1) se += __shfl_xor(se, d, 32);
      float inv = 1.f / se;
#pragma unroll
      for (int nt = 0; nt < 4; nt++)
        ldsP[h][m * Nv + nt * 16 + ln] = f2bf(vals[nt] * inv);
    }
  }
  // P @ V  (M=64, N=32, K=64); same-wave LDS ops are in-order.
  for (int mt = 0; mt < 4; mt++)
#pragma unroll
    for (int dt = 0; dt < 2; dt++) {
      f32x8 o = {};
#pragma unroll
      for (int kt = 0; kt < 2; kt++) {
        bf16x16 a = load_frag(&ldsP[h][(mt * 16) * Nv + kt * 32], Nv);
        bf16x16 bfr = load_fragT(v + (size_t)(kt * 32) * 576 + dt * 16, 576);
        o = WMMA_BF16(a, bfr, o);
      }
#pragma unroll
      for (int r = 0; r < 8; r++) {
        int m = mt * 16 + mh * 8 + r;
        int d = dt * 16 + ln;
        xsp[((size_t)win * Nv + m) * Cv + h * HDv + d] = o[r];
      }
    }
}

// ---------------------------------------------------------------------------
// rfft2 of each 8x8 window per channel (ortho => 1/8), output stacked
// real rows then imag rows per window: row = win*80 + part*40 + (u*5+v).
// ---------------------------------------------------------------------------
__global__ __launch_bounds__(256) void rfft2_kernel(const float* __restrict__ xw,
                                                    bf16* __restrict__ xf) {
  int t = blockIdx.x * blockDim.x + threadIdx.x;
  if (t >= BWv * Cv) return;
  int win = t / Cv, c = t % Cv;
  float xv[64];
#pragma unroll
  for (int n = 0; n < 64; n++) xv[n] = xw[((size_t)win * Nv + n) * Cv + c];
  float tr[8][5], ti[8][5];
#pragma unroll
  for (int hh = 0; hh < 8; hh++)
#pragma unroll
    for (int vv = 0; vv < 5; vv++) {
      float sr = 0.f, si = 0.f;
#pragma unroll
      for (int w = 0; w < 8; w++) {
        int a = (vv * w) & 7;
        sr += xv[hh * 8 + w] * COS8[a];
        si -= xv[hh * 8 + w] * SIN8[a];
      }
      tr[hh][vv] = sr; ti[hh][vv] = si;
    }
#pragma unroll
  for (int u = 0; u < 8; u++)
#pragma unroll
    for (int vv = 0; vv < 5; vv++) {
      float Xr = 0.f, Xi = 0.f;
#pragma unroll
      for (int hh = 0; hh < 8; hh++) {
        int a = (u * hh) & 7;
        float cu = COS8[a], su = SIN8[a];
        Xr += tr[hh][vv] * cu + ti[hh][vv] * su;   // (tr+i ti)*(cu - i su)
        Xi += ti[hh][vv] * cu - tr[hh][vv] * su;
      }
      size_t rrow = (size_t)win * 80 + u * 5 + vv;
      xf[rrow * Cv + c]              = f2bf(Xr * 0.125f);
      xf[(rrow + 40) * Cv + c]       = f2bf(Xi * 0.125f);
    }
}

// ---------------------------------------------------------------------------
// Fourier attention: one block per (window, head).  40 freq tokens, hd=32.
// attn = |q k^T| * scale (no conj, matches reference), softmax, @ complex v.
// ---------------------------------------------------------------------------
__global__ __launch_bounds__(256) void fourier_attn(
    const bf16* __restrict__ qkvf, float* __restrict__ xfr, float* __restrict__ xfi) {
  int win = blockIdx.x / HEADSv, h = blockIdx.x % HEADSv;
  __shared__ float sq[2][40][32], sk[2][40][32], sv[2][40][32];
  __shared__ float att[40][40];
  int t = threadIdx.x;
  for (int i = t; i < 40 * 32; i += 256) {
    int f = i >> 5, d = i & 31;
    size_t base = ((size_t)win * 80 + f) * 576 + h * HDv + d;
    sq[0][f][d] = (float)qkvf[base];
    sq[1][f][d] = (float)qkvf[base + (size_t)40 * 576];
    sk[0][f][d] = (float)qkvf[base + Cv];
    sk[1][f][d] = (float)qkvf[base + (size_t)40 * 576 + Cv];
    sv[0][f][d] = (float)qkvf[base + 2 * Cv];
    sv[1][f][d] = (float)qkvf[base + (size_t)40 * 576 + 2 * Cv];
  }
  __syncthreads();
  for (int e = t; e < 1600; e += 256) {
    int n = e / 40, m2 = e % 40;
    float ar = 0.f, ai = 0.f;
    for (int d = 0; d < 32; d++) {
      float qr = sq[0][n][d], qi = sq[1][n][d];
      float kr = sk[0][m2][d], ki = sk[1][m2][d];
      ar += qr * kr - qi * ki;
      ai += qr * ki + qi * kr;
    }
    att[n][m2] = SCALEv * sqrtf(ar * ar + ai * ai);
  }
  __syncthreads();
  if (t < 40) {
    float mx = -1e30f;
    for (int m2 = 0; m2 < 40; m2++) mx = fmaxf(mx, att[t][m2]);
    float s = 0.f;
    for (int m2 = 0; m2 < 40; m2++) { float e = __expf(att[t][m2] - mx); att[t][m2] = e; s += e; }
    float inv = 1.f / s;
    for (int m2 = 0; m2 < 40; m2++) att[t][m2] *= inv;
  }
  __syncthreads();
  for (int e = t; e < 40 * 32; e += 256) {
    int n = e >> 5, d = e & 31;
    float orr = 0.f, oi = 0.f;
    for (int m2 = 0; m2 < 40; m2++) {
      float p = att[n][m2];
      orr += p * sv[0][m2][d];
      oi  += p * sv[1][m2][d];
    }
    size_t o = ((size_t)win * 40 + n) * Cv + h * HDv + d;
    xfr[o] = orr; xfi[o] = oi;
  }
}

// ---------------------------------------------------------------------------
// irfft2 (s=(8,8), ortho): hermitian mirror Xfull[u][v>=5] = conj(X[(8-u)&7][8-v]).
// ---------------------------------------------------------------------------
__global__ __launch_bounds__(256) void irfft2_kernel(
    const float* __restrict__ xfr, const float* __restrict__ xfi,
    float* __restrict__ xfour) {
  int t = blockIdx.x * blockDim.x + threadIdx.x;
  if (t >= BWv * Cv) return;
  int win = t / Cv, c = t % Cv;
  float Xr[8][5], Xi[8][5];
#pragma unroll
  for (int u = 0; u < 8; u++)
#pragma unroll
    for (int vv = 0; vv < 5; vv++) {
      size_t idx = ((size_t)win * 40 + u * 5 + vv) * Cv + c;
      Xr[u][vv] = xfr[idx]; Xi[u][vv] = xfi[idx];
    }
  for (int hh = 0; hh < 8; hh++)
    for (int w = 0; w < 8; w++) {
      float s = 0.f;
#pragma unroll
      for (int u = 0; u < 8; u++)
#pragma unroll
        for (int vv = 0; vv < 8; vv++) {
          float ar, ai;
          if (vv < 5) { ar = Xr[u][vv]; ai = Xi[u][vv]; }
          else { int uu = (8 - u) & 7; ar = Xr[uu][8 - vv]; ai = -Xi[uu][8 - vv]; }
          int a = (u * hh + vv * w) & 7;
          s += ar * COS8[a] - ai * SIN8[a];
        }
      xfour[((size_t)win * Nv + hh * 8 + w) * Cv + c] = s * 0.125f;
    }
}

__global__ void combine_kernel(const float* __restrict__ a, const float* __restrict__ b,
                               bf16* __restrict__ o, size_t n) {
  size_t i = (size_t)blockIdx.x * blockDim.x + threadIdx.x;
  if (i < n) o[i] = f2bf(a[i] + b[i]);
}

// ---------------------------------------------------------------------------
extern "C" void kernel_launch(void* const* d_in, const int* in_sizes, int n_in,
                              void* d_out, int out_size, void* d_ws, size_t ws_size,
                              hipStream_t stream) {
  const float* x      = (const float*)d_in[0];
  const float* mask   = (const float*)d_in[1];
  const float* n1g    = (const float*)d_in[2];
  const float* n1b    = (const float*)d_in[3];
  const float* qkv_w  = (const float*)d_in[4];
  const float* qkv_b  = (const float*)d_in[5];
  const float* relt   = (const float*)d_in[6];
  const float* proj_w = (const float*)d_in[7];
  const float* proj_b = (const float*)d_in[8];
  const float* n2g    = (const float*)d_in[9];
  const float* n2b    = (const float*)d_in[10];
  const float* fc1_w  = (const float*)d_in[11];
  const float* fc1_b  = (const float*)d_in[12];
  const float* fc2_w  = (const float*)d_in[13];
  const float* fc2_b  = (const float*)d_in[14];
  float* out = (float*)d_out;

  char* ws = (char*)d_ws;
  size_t off = 0;
  auto alloc = [&](size_t bytes) -> void* {
    void* p = ws + off;
    off += (bytes + 255) & ~(size_t)255;
    return p;
  };
  float* XW_F32  = (float*)alloc((size_t)MTOK * Cv * 4);
  bf16*  XW_BF   = (bf16*) alloc((size_t)MTOK * Cv * 2);
  bf16*  QKV_BF  = (bf16*) alloc((size_t)MTOK * 576 * 2);
  float* XSP     = (float*)alloc((size_t)MTOK * Cv * 4);
  bf16*  XF_BF   = (bf16*) alloc((size_t)MFOUR * Cv * 2);
  bf16*  QKVF_BF = (bf16*) alloc((size_t)MFOUR * 576 * 2);
  float* XFR     = (float*)alloc((size_t)BWv * 40 * Cv * 4);
  float* XFI     = (float*)alloc((size_t)BWv * 40 * Cv * 4);
  float* XFOUR   = (float*)alloc((size_t)MTOK * Cv * 4);
  bf16*  AIN_BF  = (bf16*) alloc((size_t)MTOK * Cv * 2);
  float* X1      = (float*)alloc((size_t)MTOK * Cv * 4);
  bf16*  YN_BF   = (bf16*) alloc((size_t)MTOK * Cv * 2);
  bf16*  H_BF    = (bf16*) alloc((size_t)MTOK * 768 * 2);
  bf16*  QKVT    = (bf16*) alloc((size_t)576 * Cv * 2);
  bf16*  PROJT   = (bf16*) alloc((size_t)Cv * Cv * 2);
  bf16*  FC1T    = (bf16*) alloc((size_t)768 * Cv * 2);
  bf16*  FC2T    = (bf16*) alloc((size_t)Cv * 768 * 2);

  // 1) transpose+convert weights (stay L2-resident for all GEMM passes)
  transpose_to_bf16<<<(Cv * 576 + 255) / 256, 256, 0, stream>>>(qkv_w, QKVT, Cv, 576);
  transpose_to_bf16<<<(Cv * Cv  + 255) / 256, 256, 0, stream>>>(proj_w, PROJT, Cv, Cv);
  transpose_to_bf16<<<(Cv * 768 + 255) / 256, 256, 0, stream>>>(fc1_w, FC1T, Cv, 768);
  transpose_to_bf16<<<(768 * Cv + 255) / 256, 256, 0, stream>>>(fc2_w, FC2T, 768, Cv);

  // 2) LN1 + shift + partition
  ln_shift_partition<<<MTOK / 8, 256, 0, stream>>>(x, n1g, n1b, XW_F32, XW_BF);

  // 3) QKV GEMM  (131072 x 192) @ (192 x 576)
  gemm_bf16<0><<<dim3(MTOK / 128, 576 / 64), 256, 0, stream>>>(
      XW_BF, QKVT, qkv_b, QKV_BF, nullptr, MTOK, 576, Cv);

  // 4) spatial window attention
  spatial_attn<<<BWv, 256, 0, stream>>>(QKV_BF, mask, relt, XSP);

  // 5) rfft2 per window/channel
  rfft2_kernel<<<(BWv * Cv + 255) / 256, 256, 0, stream>>>(XW_F32, XF_BF);

  // 6) Fourier QKV GEMM  (163840 x 192) @ (192 x 576)
  gemm_bf16<0><<<dim3(MFOUR / 128, 576 / 64), 256, 0, stream>>>(
      XF_BF, QKVT, qkv_b, QKVF_BF, nullptr, MFOUR, 576, Cv);

  // 7) complex Fourier attention (40 tokens/head — VALU; pads poorly to 16)
  fourier_attn<<<BWv * HEADSv, 256, 0, stream>>>(QKVF_BF, XFR, XFI);

  // 8) irfft2
  irfft2_kernel<<<(BWv * Cv + 255) / 256, 256, 0, stream>>>(XFR, XFI, XFOUR);

  // 9) combine spatial + fourier -> bf16 proj input
  {
    size_t n = (size_t)MTOK * Cv;
    combine_kernel<<<(unsigned)((n + 255) / 256), 256, 0, stream>>>(XSP, XFOUR, AIN_BF, n);
  }

  // 10) proj GEMM + window reverse + unshift + shortcut residual -> X1 (fp32)
  gemm_bf16<2><<<dim3(MTOK / 128, Cv / 64), 256, 0, stream>>>(
      AIN_BF, PROJT, proj_b, X1, x, MTOK, Cv, Cv);

  // 11) LN2
  ln2_kernel<<<MTOK / 8, 256, 0, stream>>>(X1, n2g, n2b, YN_BF);

  // 12) fc1 + GELU  (131072 x 192) @ (192 x 768)
  gemm_bf16<1><<<dim3(MTOK / 128, 768 / 64), 256, 0, stream>>>(
      YN_BF, FC1T, fc1_b, H_BF, nullptr, MTOK, 768, Cv);

  // 13) fc2 + residual -> final output (fp32)
  gemm_bf16<3><<<dim3(MTOK / 128, Cv / 64), 256, 0, stream>>>(
      H_BF, FC2T, fc2_b, out, X1, MTOK, Cv, 768);

  (void)in_sizes; (void)n_in; (void)out_size; (void)ws_size;
}
